// RSSM_34505767256842
// MI455X (gfx1250) — matrix operand order repaired and checked by reading, not compiled, not run on hardware
//
#include <hip/hip_runtime.h>
#include <hip/hip_bf16.h>
#include <math.h>

// ---------------------------------------------------------------------------
// RSSM rollout for MI455X (gfx1250): bf16 WMMA GEMMs, f32 accumulation.
// Weights converted once to bf16 (~20MB, L2-resident). Output stores are
// non-temporal (write-once 201MB stream) to preserve L2 for weights.
// ---------------------------------------------------------------------------

typedef __attribute__((ext_vector_type(16))) __bf16 v16bf;
typedef __attribute__((ext_vector_type(8)))  __bf16 v8bf;
typedef __attribute__((ext_vector_type(8)))  float  v8f;

#define T_STEPS 64
#define BATCH   512
#define DET     1024
#define STO     128
#define EMB     1024
#define ACT     16
#define MLP     1024
#define KIN     160            // 144 (STO+ACT) padded to multiple of 32
#define OUTW    1536           // 4*STO + DET per (t,b)

__device__ __forceinline__ float sigmoidf_(float x) { return 1.0f / (1.0f + __expf(-x)); }
__device__ __forceinline__ float softplusf_(float x) { return (x > 30.0f) ? x : log1pf(__expf(x)); }
__device__ __forceinline__ float eluf_(float x) { return (x > 0.0f) ? x : (__expf(x) - 1.0f); }

// ---------------------------------------------------------------------------
// Generic GEMM: C[M,N] = A[M,K] @ W[N,K]^T (+bias), M=BATCH fixed.
// W row-major [N,K] => WMMA B-fragment is a contiguous 32B read per lane.
// Wave tile 16x64 (4 wmma tiles), block = 8 waves => 64x128 block tile.
// mode 0: Cf[m*N+n] = acc + bias    (f32)
// mode 1: Cb[m*N+n] = bf16(elu(acc + bias))
// Requires: K % 32 == 0, N % 128 == 0 or N == 256, lda % 8 == 0.
// ---------------------------------------------------------------------------
__global__ __launch_bounds__(256)
void rssm_gemm_bf16_wmma(const unsigned short* __restrict__ Ap, int lda,
                         const unsigned short* __restrict__ Wp, int K,
                         const float* __restrict__ bias,
                         float* __restrict__ Cf,
                         unsigned short* __restrict__ Cbp,
                         int N, int mode)
{
    const __bf16* A = (const __bf16*)Ap;
    const __bf16* W = (const __bf16*)Wp;
    __bf16*      Cb = (__bf16*)Cbp;

    const int tid  = threadIdx.x;
    const int lane = tid & 31;
    const int wid  = tid >> 5;
    const int wm   = wid & 3;        // 4 waves along M
    const int wn   = wid >> 2;       // 2 waves along N
    const int lr   = lane & 15;      // A: row in tile, B/C: column in tile
    const int hi   = lane >> 4;      // lane-group select (0/1)

    const int m0 = blockIdx.y * 64 + wm * 16;
    const int n0 = blockIdx.x * 128 + wn * 64;

    // A fragment base: lanes0-15 halves0-15 = K 0..15;
    //                  lanes16-31 shift K by +16.
    const __bf16* aptr = A + (size_t)(m0 + lr) * lda + (hi << 4);
    // B fragment base: lane column n, 16 contiguous K halves; lanes16-31 K+16.
    const __bf16* wptr = W + (size_t)(n0 + lr) * K + (hi << 4);
    const size_t wstep16 = (size_t)16 * K;      // advance 16 rows of W (one N tile)

    v8f acc[4] = {};

    for (int kb = 0; kb < K; kb += 32) {
        v8bf alo = *(const v8bf*)(aptr + kb);
        v8bf ahi = *(const v8bf*)(aptr + kb + 8);
        v16bf af = __builtin_shufflevector(alo, ahi,
                                           0, 1, 2, 3, 4, 5, 6, 7,
                                           8, 9, 10, 11, 12, 13, 14, 15);
#pragma unroll
        for (int t = 0; t < 4; ++t) {
            v16bf bfr = *(const v16bf*)(wptr + (size_t)t * wstep16 + kb);
            acc[t] = __builtin_amdgcn_wmma_f32_16x16x32_bf16(
                false, af, false, bfr, (short)0, acc[t], false, false);
        }
    }

    // C/D layout: VGPR i -> row m0 + i + 8*hi ; lane%16 -> column.
#pragma unroll
    for (int t = 0; t < 4; ++t) {
        const int n = n0 + t * 16 + lr;
        const float bn = bias ? bias[n] : 0.0f;
#pragma unroll
        for (int i = 0; i < 8; ++i) {
            const int m = m0 + (hi << 3) + i;
            float v = acc[t][i] + bn;
            if (mode == 0) {
                Cf[(size_t)m * N + n] = v;
            } else {
                Cb[(size_t)m * N + n] = (__bf16)eluf_(v);
            }
        }
    }
}

// ---------------------------------------------------------------------------
// Weight conversion f32 -> bf16
// ---------------------------------------------------------------------------
__global__ __launch_bounds__(256)
void rssm_convert_bf16(const float* __restrict__ src, unsigned short* __restrict__ dstp, int n)
{
    int i = blockIdx.x * 256 + threadIdx.x;
    if (i < n) ((__bf16*)dstp)[i] = (__bf16)src[i];
}

// W_in [1024,144] -> bf16 [1024,160] zero-padded along K
__global__ __launch_bounds__(256)
void rssm_convert_win(const float* __restrict__ src, unsigned short* __restrict__ dstp)
{
    int i = blockIdx.x * 256 + threadIdx.x;
    if (i >= DET * KIN) return;
    int n = i / KIN, k = i - n * KIN;
    float v = (k < (STO + ACT)) ? src[n * (STO + ACT) + k] : 0.0f;
    ((__bf16*)dstp)[i] = (__bf16)v;
}

// ---------------------------------------------------------------------------
// Step prologue: xcat = [stoc*nt, a, 0pad] (bf16), hprev = deter*nt (bf16+f32)
// ---------------------------------------------------------------------------
__global__ __launch_bounds__(256)
void rssm_prep(const float* __restrict__ stoc, const float* __restrict__ deter,
               const float* __restrict__ act_t, const float* __restrict__ nt_t,
               unsigned short* __restrict__ xcatp, unsigned short* __restrict__ hprev_bp,
               float* __restrict__ hprev_f)
{
    int idx = blockIdx.x * 256 + threadIdx.x;
    if (idx >= BATCH * (DET + KIN)) return;
    int b = idx / (DET + KIN);
    int j = idx - b * (DET + KIN);
    float nt = nt_t[b];
    if (j < DET) {
        float v = deter[b * DET + j] * nt;
        hprev_f[b * DET + j] = v;
        ((__bf16*)hprev_bp)[b * DET + j] = (__bf16)v;
    } else {
        int k = j - DET;
        float v = 0.0f;
        if (k < STO)            v = stoc[b * STO + k] * nt;
        else if (k < STO + ACT) v = act_t[b * ACT + (k - STO)];
        ((__bf16*)xcatp)[b * KIN + k] = (__bf16)v;
    }
}

// ---------------------------------------------------------------------------
// GRU gate math; writes h (f32), [h|obs] (bf16) and h into out[t] (NT stores)
// ---------------------------------------------------------------------------
__global__ __launch_bounds__(256)
void rssm_gru(const float* __restrict__ gi, const float* __restrict__ gh,
              const float* __restrict__ hprev_f, const float* __restrict__ obs_t,
              float* __restrict__ h_f, unsigned short* __restrict__ hobsp,
              float* __restrict__ out_t)
{
    int idx = blockIdx.x * 256 + threadIdx.x;
    if (idx >= BATCH * DET) return;
    int b = idx / DET, j = idx - b * DET;
    const float* gib = gi + (size_t)b * 3 * DET;
    const float* ghb = gh + (size_t)b * 3 * DET;
    float r = sigmoidf_(gib[j] + ghb[j]);
    float z = sigmoidf_(gib[DET + j] + ghb[DET + j]);
    float n = tanhf(gib[2 * DET + j] + r * ghb[2 * DET + j]);
    float h = (1.0f - z) * n + z * hprev_f[idx];
    h_f[idx] = h;
    __bf16* hb = (__bf16*)hobsp + (size_t)b * (DET + EMB);
    hb[j] = (__bf16)h;
    hb[DET + j] = (__bf16)obs_t[idx];
    __builtin_nontemporal_store(h, out_t + (size_t)b * OUTW + 4 * STO + j);
}

// ---------------------------------------------------------------------------
// Heads: softplus + MIN_STD; posterior also produces stoc = qm + qs*eps
// ---------------------------------------------------------------------------
__global__ __launch_bounds__(256)
void rssm_prior_head(const float* __restrict__ pr, float* __restrict__ out_t)
{
    int idx = blockIdx.x * 256 + threadIdx.x;
    if (idx >= BATCH * STO) return;
    int b = idx / STO, j = idx - b * STO;
    float pm = pr[b * 2 * STO + j];
    float ps = softplusf_(pr[b * 2 * STO + STO + j]) + 0.1f;
    __builtin_nontemporal_store(pm, out_t + (size_t)b * OUTW + j);
    __builtin_nontemporal_store(ps, out_t + (size_t)b * OUTW + STO + j);
}

__global__ __launch_bounds__(256)
void rssm_post_head(const float* __restrict__ po, const float* __restrict__ noise_t,
                    float* __restrict__ out_t, float* __restrict__ stoc)
{
    int idx = blockIdx.x * 256 + threadIdx.x;
    if (idx >= BATCH * STO) return;
    int b = idx / STO, j = idx - b * STO;
    float qm = po[b * 2 * STO + j];
    float qs = softplusf_(po[b * 2 * STO + STO + j]) + 0.1f;
    __builtin_nontemporal_store(qm, out_t + (size_t)b * OUTW + 2 * STO + j);
    __builtin_nontemporal_store(qs, out_t + (size_t)b * OUTW + 3 * STO + j);
    stoc[b * STO + j] = qm + qs * noise_t[b * STO + j];
}

// ---------------------------------------------------------------------------
// Host launcher
// ---------------------------------------------------------------------------
extern "C" void kernel_launch(void* const* d_in, const int* in_sizes, int n_in,
                              void* d_out, int out_size, void* d_ws, size_t ws_size,
                              hipStream_t stream)
{
    const float* actions    = (const float*)d_in[0];
    const float* nonterm    = (const float*)d_in[1];
    const float* obs        = (const float*)d_in[2];
    const float* noise      = (const float*)d_in[3];
    const float* init_stoc  = (const float*)d_in[4];
    const float* init_deter = (const float*)d_in[5];
    const float* W_in = (const float*)d_in[6];
    const float* b_in = (const float*)d_in[7];
    const float* W_ih = (const float*)d_in[8];
    const float* W_hh = (const float*)d_in[9];
    const float* b_ih = (const float*)d_in[10];
    const float* b_hh = (const float*)d_in[11];
    const float* Wp1  = (const float*)d_in[12];
    const float* bp1  = (const float*)d_in[13];
    const float* Wp2  = (const float*)d_in[14];
    const float* bp2  = (const float*)d_in[15];
    const float* Wq1  = (const float*)d_in[16];
    const float* bq1  = (const float*)d_in[17];
    const float* Wq2  = (const float*)d_in[18];
    const float* bq2  = (const float*)d_in[19];
    float* out = (float*)d_out;

    char* ws = (char*)d_ws;
    size_t off = 0;
    auto alloc = [&](size_t bytes) -> void* {
        void* p = ws + off;
        off += (bytes + 255) & ~(size_t)255;
        return p;
    };

    // bf16 weights (L2-resident across all 64 steps)
    unsigned short* wWin = (unsigned short*)alloc((size_t)DET * KIN * 2);
    unsigned short* wWih = (unsigned short*)alloc((size_t)3 * DET * DET * 2);
    unsigned short* wWhh = (unsigned short*)alloc((size_t)3 * DET * DET * 2);
    unsigned short* wWp1 = (unsigned short*)alloc((size_t)MLP * DET * 2);
    unsigned short* wWp2 = (unsigned short*)alloc((size_t)2 * STO * MLP * 2);
    unsigned short* wWq1 = (unsigned short*)alloc((size_t)MLP * (DET + EMB) * 2);
    unsigned short* wWq2 = (unsigned short*)alloc((size_t)2 * STO * MLP * 2);
    // activations
    unsigned short* xcat   = (unsigned short*)alloc((size_t)BATCH * KIN * 2);
    unsigned short* hprevb = (unsigned short*)alloc((size_t)BATCH * DET * 2);
    float*          hprevf = (float*)alloc((size_t)BATCH * DET * 4);
    unsigned short* x_bf   = (unsigned short*)alloc((size_t)BATCH * DET * 2);
    float*          gi     = (float*)alloc((size_t)BATCH * 3 * DET * 4);
    float*          gh     = (float*)alloc((size_t)BATCH * 3 * DET * 4);
    float*          h_f    = (float*)alloc((size_t)BATCH * DET * 4);
    unsigned short* hobs   = (unsigned short*)alloc((size_t)BATCH * (DET + EMB) * 2);
    unsigned short* p1     = (unsigned short*)alloc((size_t)BATCH * MLP * 2);
    unsigned short* q1     = (unsigned short*)alloc((size_t)BATCH * MLP * 2);
    float*          pr     = (float*)alloc((size_t)BATCH * 2 * STO * 4);
    float*          po     = (float*)alloc((size_t)BATCH * 2 * STO * 4);
    float*          stoc   = (float*)alloc((size_t)BATCH * STO * 4);

    // --- weight conversion (deterministic, every call) ---
    rssm_convert_win<<<(DET * KIN + 255) / 256, 256, 0, stream>>>(W_in, wWin);
    rssm_convert_bf16<<<(3 * DET * DET + 255) / 256, 256, 0, stream>>>(W_ih, wWih, 3 * DET * DET);
    rssm_convert_bf16<<<(3 * DET * DET + 255) / 256, 256, 0, stream>>>(W_hh, wWhh, 3 * DET * DET);
    rssm_convert_bf16<<<(MLP * DET + 255) / 256, 256, 0, stream>>>(Wp1, wWp1, MLP * DET);
    rssm_convert_bf16<<<(2 * STO * MLP + 255) / 256, 256, 0, stream>>>(Wp2, wWp2, 2 * STO * MLP);
    rssm_convert_bf16<<<(MLP * (DET + EMB) + 255) / 256, 256, 0, stream>>>(Wq1, wWq1, MLP * (DET + EMB));
    rssm_convert_bf16<<<(2 * STO * MLP + 255) / 256, 256, 0, stream>>>(Wq2, wWq2, 2 * STO * MLP);

    // --- init recurrent state ---
    hipMemcpyAsync(stoc, init_stoc, (size_t)BATCH * STO * 4, hipMemcpyDeviceToDevice, stream);
    hipMemcpyAsync(h_f, init_deter, (size_t)BATCH * DET * 4, hipMemcpyDeviceToDevice, stream);

    const dim3 blk(256);
    const dim3 gN1024(DET / 128, BATCH / 64);       // N=1024
    const dim3 gN3072(3 * DET / 128, BATCH / 64);   // N=3072
    const dim3 gN256(2, BATCH / 64);                // N=256

    for (int t = 0; t < T_STEPS; ++t) {
        const float* act_t   = actions + (size_t)t * BATCH * ACT;
        const float* nt_t    = nonterm + (size_t)t * BATCH;
        const float* obs_t   = obs + (size_t)t * BATCH * EMB;
        const float* noise_t = noise + (size_t)t * BATCH * STO;
        float* out_t = out + (size_t)t * BATCH * OUTW;

        rssm_prep<<<(BATCH * (DET + KIN) + 255) / 256, blk, 0, stream>>>(
            stoc, h_f, act_t, nt_t, xcat, hprevb, hprevf);

        // x = elu(xcat @ W_in^T + b_in)
        rssm_gemm_bf16_wmma<<<gN1024, blk, 0, stream>>>(
            xcat, KIN, wWin, KIN, b_in, nullptr, x_bf, DET, 1);
        // gi = x @ W_ih^T + b_ih ; gh = hprev @ W_hh^T + b_hh
        rssm_gemm_bf16_wmma<<<gN3072, blk, 0, stream>>>(
            x_bf, DET, wWih, DET, b_ih, gi, nullptr, 3 * DET, 0);
        rssm_gemm_bf16_wmma<<<gN3072, blk, 0, stream>>>(
            hprevb, DET, wWhh, DET, b_hh, gh, nullptr, 3 * DET, 0);

        rssm_gru<<<(BATCH * DET + 255) / 256, blk, 0, stream>>>(
            gi, gh, hprevf, obs_t, h_f, hobs, out_t);

        // prior: p1 = elu(h @ Wp1^T + bp1); pr = p1 @ Wp2^T + bp2
        rssm_gemm_bf16_wmma<<<gN1024, blk, 0, stream>>>(
            hobs, DET + EMB, wWp1, DET, bp1, nullptr, p1, MLP, 1);
        rssm_gemm_bf16_wmma<<<gN256, blk, 0, stream>>>(
            p1, MLP, wWp2, MLP, bp2, pr, nullptr, 2 * STO, 0);
        // posterior: q1 = elu([h,obs] @ Wq1^T + bq1); po = q1 @ Wq2^T + bq2
        rssm_gemm_bf16_wmma<<<gN1024, blk, 0, stream>>>(
            hobs, DET + EMB, wWq1, DET + EMB, bq1, nullptr, q1, MLP, 1);
        rssm_gemm_bf16_wmma<<<gN256, blk, 0, stream>>>(
            q1, MLP, wWq2, MLP, bq2, po, nullptr, 2 * STO, 0);

        rssm_prior_head<<<(BATCH * STO + 255) / 256, blk, 0, stream>>>(pr, out_t);
        rssm_post_head<<<(BATCH * STO + 255) / 256, blk, 0, stream>>>(po, noise_t, out_t, stoc);
    }
}
